// SpaDOT_49538152792510
// MI455X (gfx1250) — compile-verified, matrix operand order checked by reading, not run on hardware
//
#include <hip/hip_runtime.h>
#include <math.h>

#define LOG2PI_F 1.8378770664093453f

typedef __bf16 bf16;
typedef __attribute__((ext_vector_type(16))) __bf16 v16bf;
typedef __attribute__((ext_vector_type(8)))  float  v8f;

static __device__ __forceinline__ bf16 f2bf(float f) {
  unsigned u = __float_as_uint(f);
  unsigned short h = (unsigned short)(u >> 16);
  bf16 r;
  __builtin_memcpy(&r, &h, 2);
  return r;
}

static __device__ __forceinline__ float eluf(float x) {
  return x > 0.f ? x : (expf(x) - 1.f);
}

static __device__ void atomicMaxF(float* addr, float val) {
  unsigned* ua = (unsigned*)addr;
  unsigned old = *ua;
  while (true) {
    float cur = __uint_as_float(old);
    if (cur >= val) break;
    unsigned assumed = old;
    old = atomicCAS(ua, assumed, __float_as_uint(val));
    if (old == assumed) break;
  }
}

static __device__ float hashnorm(unsigned seed, unsigned idx) {
  unsigned x = idx * 2654435761u ^ (seed * 374761393u + 0x9e3779b9u);
  x ^= x >> 13; x *= 1274126177u; x ^= x >> 16;
  unsigned y = (x + idx * 668265263u) * 2246822519u;
  y ^= y >> 13; y *= 3266489917u; y ^= y >> 16;
  float u1 = ((x & 0xFFFFFFu) + 1u) * (1.0f / 16777216.0f);
  float u2 = (y & 0xFFFFFFu) * (1.0f / 16777216.0f);
  return sqrtf(-2.f * logf(u1)) * cosf(6.28318530718f * u2);
}

// ---------------------------------------------------------------------------
// Pack a KxH row-major f32 weight matrix into bf16 WMMA-B-fragment order:
// P[(((chunk*Hpad + col)*2 + kbsel)*16) + t] where for kbsel (0: lanes 0-15,
// 1: lanes 16-31, kb = 8*kbsel): k = 32*chunk + (t<8 ? kb+t : 16+kb+t-8).
// Zero-fills k>=K and col>=H so the GEMM hot loop needs NO guards.
// ---------------------------------------------------------------------------
__global__ void packB(const float* __restrict__ B, bf16* __restrict__ P,
                      int K, int H, int Hpad, int Kc) {
  long idx = (long)blockIdx.x * blockDim.x + threadIdx.x;
  long tot = (long)Kc * Hpad * 32;
  if (idx >= tot) return;
  int t = (int)(idx & 15);
  int v = (int)((idx >> 4) & 1);
  long cj = idx >> 5;
  int j = (int)(cj % Hpad);
  int ch = (int)(cj / Hpad);
  int kb = v * 8;
  int k = ch * 32 + (t < 8 ? kb + t : 16 + kb + (t - 8));
  float val = (k < K && j < H) ? B[(size_t)k * H + j] : 0.f;
  P[idx] = f2bf(val);
}

static __device__ __forceinline__ v16bf loadA_f32x8x2(const float* p0, const float* p1) {
  float4 a0 = *(const float4*)p0;
  float4 a1 = *(const float4*)(p0 + 4);
  float4 a2 = *(const float4*)p1;
  float4 a3 = *(const float4*)(p1 + 4);
  v16bf a;
  a[0] = f2bf(a0.x); a[1] = f2bf(a0.y); a[2] = f2bf(a0.z); a[3] = f2bf(a0.w);
  a[4] = f2bf(a1.x); a[5] = f2bf(a1.y); a[6] = f2bf(a1.z); a[7] = f2bf(a1.w);
  a[8] = f2bf(a2.x); a[9] = f2bf(a2.y); a[10] = f2bf(a2.z); a[11] = f2bf(a2.w);
  a[12] = f2bf(a3.x); a[13] = f2bf(a3.y); a[14] = f2bf(a3.z); a[15] = f2bf(a3.w);
  return a;
}

// ---------------------------------------------------------------------------
// Fused dual-B WMMA GEMM: C1 = A@B1 (raw), C2 = elu(A@B2 + bias2).
// A (f32) streamed once via aligned float4 loads; B pre-packed bf16 fragments
// (one v16bf load each). Main K loop is branch-free (tail split out);
// unroll-2 lets the scheduler hide B-load latency under the WMMAs.
// ---------------------------------------------------------------------------
__global__ __launch_bounds__(32) void gemm_dualB_wmma(
    const float* __restrict__ A, const bf16* __restrict__ P1,
    const bf16* __restrict__ P2, const float* __restrict__ bias2,
    float* __restrict__ C1, float* __restrict__ C2,
    int N, int K, int H, int Kc) {
  int lane = threadIdx.x;
  int row0 = blockIdx.x * 16;
  int col0 = blockIdx.y * 32;
  int arow = row0 + (lane & 15);
  int kb = (lane < 16) ? 0 : 8;
  int vsel = (lane < 16) ? 0 : 1;
  int ca = col0 + (lane & 15);
  int cb = ca + 16;
  const float* Arow = A + (size_t)arow * K;
  v8f c1a = {}, c1b = {}, c2a = {}, c2b = {};
  int KcFull = K / 32;
#pragma unroll 2
  for (int ch = 0; ch < KcFull; ++ch) {
    int k0 = ch * 32;
    __builtin_prefetch(Arow + k0 + 128, 0, 1);  // global_prefetch_b8
    v16bf a = loadA_f32x8x2(Arow + k0 + kb, Arow + k0 + 16 + kb);
    size_t base_a = (((size_t)ch * H + ca) * 2 + vsel) * 16;
    size_t base_b = (((size_t)ch * H + cb) * 2 + vsel) * 16;
    v16bf b1a = *(const v16bf*)(P1 + base_a);
    v16bf b1b = *(const v16bf*)(P1 + base_b);
    v16bf b2a = *(const v16bf*)(P2 + base_a);
    v16bf b2b = *(const v16bf*)(P2 + base_b);
    c1a = __builtin_amdgcn_wmma_f32_16x16x32_bf16(false, a, false, b1a, (short)0, c1a, false, false);
    c1b = __builtin_amdgcn_wmma_f32_16x16x32_bf16(false, a, false, b1b, (short)0, c1b, false, false);
    c2a = __builtin_amdgcn_wmma_f32_16x16x32_bf16(false, a, false, b2a, (short)0, c2a, false, false);
    c2b = __builtin_amdgcn_wmma_f32_16x16x32_bf16(false, a, false, b2b, (short)0, c2b, false, false);
  }
  if (KcFull < Kc) {  // single guarded tail chunk (K % 32 != 0)
    int ch = KcFull;
    int k0 = ch * 32;
    v16bf a;
#pragma unroll
    for (int t = 0; t < 8; t++) {
      int kA = k0 + kb + t;
      int kB = k0 + 16 + kb + t;
      a[t]     = f2bf(kA < K ? Arow[kA] : 0.f);
      a[t + 8] = f2bf(kB < K ? Arow[kB] : 0.f);
    }
    size_t base_a = (((size_t)ch * H + ca) * 2 + vsel) * 16;
    size_t base_b = (((size_t)ch * H + cb) * 2 + vsel) * 16;
    v16bf b1a = *(const v16bf*)(P1 + base_a);
    v16bf b1b = *(const v16bf*)(P1 + base_b);
    v16bf b2a = *(const v16bf*)(P2 + base_a);
    v16bf b2b = *(const v16bf*)(P2 + base_b);
    c1a = __builtin_amdgcn_wmma_f32_16x16x32_bf16(false, a, false, b1a, (short)0, c1a, false, false);
    c1b = __builtin_amdgcn_wmma_f32_16x16x32_bf16(false, a, false, b1b, (short)0, c1b, false, false);
    c2a = __builtin_amdgcn_wmma_f32_16x16x32_bf16(false, a, false, b2a, (short)0, c2a, false, false);
    c2b = __builtin_amdgcn_wmma_f32_16x16x32_bf16(false, a, false, b2b, (short)0, c2b, false, false);
  }
#pragma unroll
  for (int i = 0; i < 8; i++) {
    int r = row0 + ((lane >> 4) * 8) + i;
    C1[(size_t)r * H + ca] = c1a[i];
    C1[(size_t)r * H + cb] = c1b[i];
    float h1 = c2a[i] + bias2[ca];
    float h2 = c2b[i] + bias2[cb];
    C2[(size_t)r * H + ca] = eluf(h1);
    C2[(size_t)r * H + cb] = eluf(h2);
  }
}

// ---------------------------------------------------------------------------
// Decoder WMMA GEMM fused with squared-error reduction:
// accumulates sum((Y - (Hd@W2 + b2))^2) without materializing recon_x (49MB).
// K=128 (multiple of 32) -> fully unguarded hot loop; W2 pre-packed to Dpad.
// ---------------------------------------------------------------------------
__global__ __launch_bounds__(32) void gemm_dec_wmma(
    const float* __restrict__ Hd, const bf16* __restrict__ P,
    const float* __restrict__ b2, const float* __restrict__ Y,
    float* __restrict__ loss_acc, int N, int K, int D, int Dpad) {
  int lane = threadIdx.x;
  int row0 = blockIdx.x * 16;
  int col0 = blockIdx.y * 16;
  int arow = row0 + (lane & 15);
  int kb = (lane < 16) ? 0 : 8;
  int vsel = (lane < 16) ? 0 : 1;
  int cc = col0 + (lane & 15);
  const float* Arow = Hd + (size_t)arow * K;
  v8f c = {};
  int Kc = K / 32;
#pragma unroll 2
  for (int ch = 0; ch < Kc; ++ch) {
    int k0 = ch * 32;
    v16bf a = loadA_f32x8x2(Arow + k0 + kb, Arow + k0 + 16 + kb);
    v16bf b = *(const v16bf*)(P + (((size_t)ch * Dpad + cc) * 2 + vsel) * 16);
    c = __builtin_amdgcn_wmma_f32_16x16x32_bf16(false, a, false, b, (short)0, c, false, false);
  }
  float local = 0.f;
  if (cc < D) {
    float bias = b2[cc];
#pragma unroll
    for (int i = 0; i < 8; i++) {
      int r = row0 + ((lane >> 4) * 8) + i;
      float val = c[i] + bias;
      float df = Y[(size_t)r * D + cc] - val;
      local += df * df;
    }
  }
  for (int off = 16; off > 0; off >>= 1) local += __shfl_down(local, off);
  if (lane == 0) atomicAdd(loss_acc, local);
}

// ---------------------------------------------------------------------------
// Generic small matvec with activation: C[n,j] = act(sum_k A[n,k]B[k,j]+bias)
// act: 0=none, 1=elu, 2=exp(clip(-15,15))
// ---------------------------------------------------------------------------
__global__ void matvec_act(const float* __restrict__ A, const float* __restrict__ B,
                           const float* __restrict__ bias, float* __restrict__ C,
                           int N, int K, int F, int act) {
  long idx = (long)blockIdx.x * blockDim.x + threadIdx.x;
  if (idx >= (long)N * F) return;
  int n = (int)(idx / F), j = (int)(idx % F);
  float acc = bias ? bias[j] : 0.f;
  for (int k = 0; k < K; k++) acc += A[(size_t)n * K + k] * B[(size_t)k * F + j];
  if (act == 1) acc = eluf(acc);
  else if (act == 2) acc = expf(fminf(fmaxf(acc, -15.f), 15.f));
  C[idx] = acc;
}

__global__ void fillf(float* p, float v, int n) {
  int i = blockIdx.x * blockDim.x + threadIdx.x;
  if (i < n) p[i] = v;
}

// ---------------------------- GAT kernels ----------------------------------
__global__ void gat_coef(const float* __restrict__ z, const float* __restrict__ as,
                         const float* __restrict__ ad, float* s, float* d, int N, int F) {
  int n = blockIdx.x * blockDim.x + threadIdx.x;
  if (n >= N) return;
  float a = 0.f, b = 0.f;
  for (int f = 0; f < F; f++) {
    float zv = z[(size_t)n * F + f];
    a += zv * as[f];
    b += zv * ad[f];
  }
  s[n] = a; d[n] = b;
}

__global__ void gat_edge_max(const int* __restrict__ ei, int E, int Etot,
                             const float* __restrict__ s, const float* __restrict__ d,
                             float* e, float* m) {
  int id = blockIdx.x * blockDim.x + threadIdx.x;
  if (id >= Etot) return;
  int src = id < E ? ei[id] : id - E;
  int dst = id < E ? ei[E + id] : id - E;
  float v = s[src] + d[dst];
  v = v > 0.f ? v : 0.2f * v;  // leaky_relu(0.2)
  e[id] = v;
  atomicMaxF(&m[dst], v);
}

__global__ void gat_edge_w(const int* __restrict__ ei, int E, int Etot,
                           const float* __restrict__ e, const float* __restrict__ m,
                           float* w, float* den) {
  int id = blockIdx.x * blockDim.x + threadIdx.x;
  if (id >= Etot) return;
  int dst = id < E ? ei[E + id] : id - E;
  float wv = expf(e[id] - m[dst]);
  w[id] = wv;
  atomicAdd(&den[dst], wv);
}

__global__ __launch_bounds__(64) void gat_edge_acc(const int* __restrict__ ei, int E,
                                                   const float* __restrict__ w,
                                                   const float* __restrict__ z,
                                                   float* num, int F) {
  int id = blockIdx.x;
  int src = id < E ? ei[id] : id - E;
  int dst = id < E ? ei[E + id] : id - E;
  float wv = w[id];
  for (int f = threadIdx.x; f < F; f += 64)
    atomicAdd(&num[(size_t)dst * F + f], wv * z[(size_t)src * F + f]);
}

__global__ void gat_final(const float* __restrict__ num, const float* __restrict__ den,
                          float* out, int N, int F, int act) {
  long idx = (long)blockIdx.x * blockDim.x + threadIdx.x;
  if (idx >= (long)N * F) return;
  int n = (int)(idx / F);
  float v = num[idx] / den[n];
  if (act == 1) v = eluf(v);
  else if (act == 2) v = expf(fminf(fmaxf(v, -15.f), 15.f));
  out[idx] = v;
}

// ---------------------------- SVGP kernels ---------------------------------
__global__ void kmm_build(const float* __restrict__ Z, float* K) {
  int idx = blockIdx.x * blockDim.x + threadIdx.x;
  if (idx >= 64 * 64) return;
  int i = idx / 64, j = idx % 64;
  float dx = Z[i * 2] - Z[j * 2], dy = Z[i * 2 + 1] - Z[j * 2 + 1];
  float d2 = dx * dx + dy * dy;
  K[idx] = 1.f / (1.f + d2) + (i == j ? 0.01f : 0.f);
}

__global__ void knm_build(const float* __restrict__ x, const float* __restrict__ Z,
                          float* K, int N) {
  int idx = blockIdx.x * blockDim.x + threadIdx.x;
  if (idx >= N * 64) return;
  int n = idx / 64, m = idx % 64;
  float dx = x[n * 2] - Z[m * 2], dy = x[n * 2 + 1] - Z[m * 2 + 1];
  float d2 = dx * dx + dy * dy;
  K[idx] = 1.f / (1.f + d2);
}

// Gauss-Jordan inverse + log|det| of a 64x64 matrix, single workgroup in LDS.
__global__ __launch_bounds__(128) void invert64_logdet(const float* __restrict__ A,
                                                       float* Ainv, float* logdet) {
  __shared__ float T[64][130];
  __shared__ float colp[64];
  __shared__ float ldacc;
  int c = threadIdx.x;  // 0..127
  for (int r = 0; r < 64; r++)
    T[r][c] = (c < 64) ? A[r * 64 + c] : (((c - 64) == r) ? 1.f : 0.f);
  if (c == 0) ldacc = 0.f;
  __syncthreads();
  for (int p = 0; p < 64; p++) {
    float piv = T[p][p];
    if (c == 0) ldacc += logf(fabsf(piv) + 1e-30f);
    if (c < 64) colp[c] = T[c][p];
    __syncthreads();
    float rp = T[p][c] / piv;
    T[p][c] = rp;
    __syncthreads();
    for (int r = 0; r < 64; r++)
      if (r != p) T[r][c] -= colp[r] * rp;
    __syncthreads();
  }
  for (int r = 0; r < 64; r++)
    if (c >= 64) Ainv[r * 64 + (c - 64)] = T[r][c];
  if (c == 0) *logdet = ldacc;
}

__global__ __launch_bounds__(64) void rhs_build(const float* __restrict__ Knm,
                                                const float* __restrict__ qmu,
                                                const float* __restrict__ qvar,
                                                float* rhs, int N, int l, int LZ) {
  int i = threadIdx.x;
  float acc = 0.f;
  for (int n = 0; n < N; n++)
    acc += Knm[(size_t)n * 64 + i] * qmu[(size_t)n * LZ + l] / qvar[(size_t)n * LZ + l];
  rhs[i] = acc;
}

__global__ __launch_bounds__(64) void sigma_build(const float* __restrict__ Kmm,
                                                  const float* __restrict__ Knm,
                                                  const float* __restrict__ qvar,
                                                  float* sigma, int N, int l, int LZ) {
  int i = blockIdx.x, j = threadIdx.x;
  float acc = 0.f;
  for (int n = 0; n < N; n++) {
    float p = 1.f / qvar[(size_t)n * LZ + l];
    acc += Knm[(size_t)n * 64 + i] * Knm[(size_t)n * 64 + j] * p;
  }
  sigma[i * 64 + j] = Kmm[i * 64 + j] + acc;
}

__global__ __launch_bounds__(64) void matvec64(const float* __restrict__ M,
                                               const float* __restrict__ v, float* out) {
  int i = threadIdx.x;
  float a = 0.f;
  for (int j = 0; j < 64; j++) a += M[i * 64 + j] * v[j];
  out[i] = a;
}

// kl = 0.5*(logdet_sigma - logdet_Kmm - 64 + tr(sigma_inv@Kmm) + mu_hat.tmp1)
__global__ __launch_bounds__(64) void kl_kernel(const float* __restrict__ siginv,
                                                const float* __restrict__ Kmm,
                                                const float* __restrict__ muhat,
                                                const float* __restrict__ tmp1,
                                                const float* ld_sigma, const float* ld_kmm,
                                                float* kl_out) {
  int i = threadIdx.x;
  __shared__ float r0[64], r1[64];
  float tr = 0.f;
  for (int k = 0; k < 64; k++) tr += siginv[i * 64 + k] * Kmm[k * 64 + i];
  r0[i] = tr;
  r1[i] = muhat[i] * tmp1[i];
  __syncthreads();
  for (int s = 32; s > 0; s >>= 1) {
    if (i < s) { r0[i] += r0[i + s]; r1[i] += r1[i + s]; }
    __syncthreads();
  }
  if (i == 0) *kl_out = 0.5f * (*ld_sigma - *ld_kmm - 64.f + r0[0] + r1[0]);
}

// Per-row SVGP: u=sigma_inv@v, w=Kmm_inv@v give p_m, p_v, k_tilde, tr, recon.
__global__ __launch_bounds__(64) void svgp_pern(const float* __restrict__ Knm,
                                                const float* __restrict__ siginv,
                                                const float* __restrict__ kinv,
                                                const float* __restrict__ tmp1,
                                                const float* __restrict__ qmu,
                                                const float* __restrict__ qvar,
                                                int l, int LZ,
                                                float* p_m, float* p_v, float* recon_acc) {
  int n = blockIdx.x, k = threadIdx.x;
  __shared__ float v[64], r0[64], r1[64], r2[64];
  v[k] = Knm[(size_t)n * 64 + k];
  __syncthreads();
  float u = 0.f, w = 0.f;
  for (int m = 0; m < 64; m++) {
    float vm = v[m];
    u += siginv[k * 64 + m] * vm;
    w += kinv[k * 64 + m] * vm;
  }
  r0[k] = v[k] * u;
  r1[k] = v[k] * w;
  r2[k] = v[k] * tmp1[k];
  __syncthreads();
  for (int s = 32; s > 0; s >>= 1) {
    if (k < s) { r0[k] += r0[k + s]; r1[k] += r1[k + s]; r2[k] += r2[k + s]; }
    __syncthreads();
  }
  if (k == 0) {
    float vu = r0[0], vw = r1[0], pm = r2[0];
    float noise = qvar[(size_t)n * LZ + l];
    float prec = 1.f / noise;
    float pv = 1.f + vu - vw;                 // Mdiff = sigma_inv - Kmm_inv
    float ktl = prec * (1.f - vw);            // k_tilde
    float tr = prec * vu;                     // Mmat == sigma_inv
    float df = qmu[(size_t)n * LZ + l] - pm;  // mean_vec == p_m
    float rc = -0.5f * (ktl + tr + LOG2PI_F + logf(noise) + prec * df * df);
    p_m[(size_t)n * LZ + l] = pm;
    p_v[(size_t)n * LZ + l] = pv;
    atomicAdd(recon_acc, rc);
  }
}

__global__ void ce_sample(const float* __restrict__ p_m, const float* __restrict__ p_v,
                          const float* __restrict__ qmu, const float* __restrict__ qvar,
                          float* latent, float* svgp_out, float* ce_acc,
                          int N, int LZ, int GZ) {
  int idx = blockIdx.x * blockDim.x + threadIdx.x;
  if (idx >= N * LZ) return;
  int n = idx / LZ, l = idx % LZ;
  float pv = fmaxf(p_v[idx], 1e-8f);
  float pm = p_m[idx];
  float qm = qmu[idx], qv = qvar[idx];
  atomicAdd(ce_acc, -0.5f * (LOG2PI_F + logf(pv) + (qv + (qm - pm) * (qm - pm)) / pv));
  float s = pm + hashnorm(43u, (unsigned)idx) * sqrtf(pv);
  svgp_out[idx] = s;
  latent[(size_t)n * (LZ + GZ) + l] = s;
}

__global__ void gnn_sample(const float* __restrict__ mu, const float* __restrict__ var,
                           float* latent, float* gnn_out, float* acc,
                           int N, int LZ, int GZ) {
  int idx = blockIdx.x * blockDim.x + threadIdx.x;
  if (idx >= N * GZ) return;
  int n = idx / GZ, g = idx % GZ;
  float m = mu[idx], v = var[idx];
  float s = m + hashnorm(42u, (unsigned)idx) * sqrtf(v);
  gnn_out[idx] = s;
  latent[(size_t)n * (LZ + GZ) + LZ + g] = s;
  atomicAdd(acc, 1.f + logf(v) - m * m - v);  // raw sum; -0.5/GZ applied in finalize
}

__global__ void align_kernel(const float* __restrict__ sv, const float* __restrict__ gn,
                             float* acc, int N, int LZ, int GZ) {
  int n = blockIdx.x * blockDim.x + threadIdx.x;
  if (n >= N) return;
  float a = 0.f, b = 0.f;
  for (int l = 0; l < LZ; l++) { float t = sv[(size_t)n * LZ + l]; a += t * t; }
  for (int g = 0; g < GZ; g++) { float t = gn[(size_t)n * GZ + g]; b += t * t; }
  float d = sqrtf(a) / (float)LZ - sqrtf(b) / (float)GZ;
  atomicAdd(acc, d * d);
}

// scal layout: 0=logdet_Kmm 1=logdet_sigma 2..6=kls 7..11=recons 12=ce 13=recon 14=gnn 15=align
__global__ void finalize(const float* __restrict__ scal, const int* __restrict__ bptr,
                         float* out, int Dd, int LZ, int GZ) {
  if (threadIdx.x != 0 || blockIdx.x != 0) return;
  float sk = 0.f, sr = 0.f;
  for (int l = 0; l < LZ; l++) { sk += scal[2 + l]; sr += scal[7 + l]; }
  float b = (float)(*bptr);
  float inside = sr - (b / 4096.0f) * sk;  // N_TRAIN = 4096
  out[0] = scal[13] / (float)Dd;
  out[1] = -fabsf(scal[12] - inside) / (float)LZ;
  out[2] = -0.5f * scal[14] / (float)GZ;
  out[3] = scal[15];
}

// ---------------------------------------------------------------------------
extern "C" void kernel_launch(void* const* d_in, const int* in_sizes, int n_in,
                              void* d_out, int out_size, void* d_ws, size_t ws_size,
                              hipStream_t stream) {
  (void)n_in; (void)out_size; (void)ws_size;
  const float* x       = (const float*)d_in[0];
  const float* y       = (const float*)d_in[1];
  const float* enc_W1  = (const float*)d_in[2];
  const float* enc_b1  = (const float*)d_in[3];
  const float* enc_Wmu = (const float*)d_in[4];
  const float* enc_bmu = (const float*)d_in[5];
  const float* enc_Wvar= (const float*)d_in[6];
  const float* enc_bvar= (const float*)d_in[7];
  const float* gat_W1  = (const float*)d_in[8];
  const float* gat_a1s = (const float*)d_in[9];
  const float* gat_a1d = (const float*)d_in[10];
  const float* gat_Wmu = (const float*)d_in[11];
  const float* gat_ams = (const float*)d_in[12];
  const float* gat_amd = (const float*)d_in[13];
  const float* gat_Wvar= (const float*)d_in[14];
  const float* gat_avs = (const float*)d_in[15];
  const float* gat_avd = (const float*)d_in[16];
  const float* dec_W1  = (const float*)d_in[17];
  const float* dec_b1  = (const float*)d_in[18];
  const float* dec_W2  = (const float*)d_in[19];
  const float* dec_b2  = (const float*)d_in[20];
  const float* Zp      = (const float*)d_in[21];
  const int*   ei      = (const int*)d_in[22];
  const int*   bptr    = (const int*)d_in[23];

  const int N  = in_sizes[0] / 2;        // 4096
  const int Dd = in_sizes[1] / N;        // 3000
  const int H  = in_sizes[3];            // 128
  const int LZ = in_sizes[5];            // 5
  const int GZ = in_sizes[12];           // 15
  const int DH = in_sizes[18];           // 128
  const int E  = in_sizes[22] / 2;       // 24576
  const int Etot = E + N;
  const int Kc1  = (Dd + 31) / 32;       // 94 chunks for the big GEMMs
  const int Dpad = (Dd + 15) & ~15;      // 3008 padded cols for decoder pack
  const int KcD  = DH / 32;              // 4 chunks for decoder

  float* wsp = (float*)d_ws;
  size_t o = 0;
  auto align16 = [&]() { o = (o + 15) & ~(size_t)15; };
  float* z1   = wsp + o; o += (size_t)N * H;
  float* hh   = wsp + o; o += (size_t)N * H;
  float* hbuf = wsp + o; o += (size_t)N * H;
  float* numb = wsp + o; o += (size_t)N * H;
  float* sb   = wsp + o; o += N;
  float* db   = wsp + o; o += N;
  float* mb   = wsp + o; o += N;
  float* denb = wsp + o; o += N;
  float* eb   = wsp + o; o += Etot;
  float* wb   = wsp + o; o += Etot;
  float* zmu  = wsp + o; o += (size_t)N * GZ;
  float* zvar = wsp + o; o += (size_t)N * GZ;
  float* gmu  = wsp + o; o += (size_t)N * GZ;
  float* gvar = wsp + o; o += (size_t)N * GZ;
  float* qmu  = wsp + o; o += (size_t)N * LZ;
  float* qvar = wsp + o; o += (size_t)N * LZ;
  float* Kmm  = wsp + o; o += 64 * 64;
  float* Kinv = wsp + o; o += 64 * 64;
  float* Knm  = wsp + o; o += (size_t)N * 64;
  float* sig  = wsp + o; o += 64 * 64;
  float* sigi = wsp + o; o += 64 * 64;
  float* rhs  = wsp + o; o += 64;
  float* tmp1 = wsp + o; o += 64;
  float* muh  = wsp + o; o += 64;
  float* pm   = wsp + o; o += (size_t)N * LZ;
  float* pv   = wsp + o; o += (size_t)N * LZ;
  float* hd   = wsp + o; o += (size_t)N * DH;
  float* scal = wsp + o; o += 16;
  // bf16 pack regions (32-byte aligned: offsets are multiples of 16 floats)
  align16();
  bf16* pB1 = (bf16*)(wsp + o); o += ((size_t)Kc1 * H * 32 + 1) / 2;
  align16();
  bf16* pB2 = (bf16*)(wsp + o); o += ((size_t)Kc1 * H * 32 + 1) / 2;
  align16();
  bf16* pBd = (bf16*)(wsp + o); o += ((size_t)KcD * Dpad * 32 + 1) / 2;

  float* out      = (float*)d_out;
  float* latent   = out + 4;
  float* svgp_out = latent + (size_t)N * (LZ + GZ);
  float* gnn_out  = svgp_out + (size_t)N * LZ;

  auto fill = [&](float* p, float v, size_t n) {
    fillf<<<dim3((unsigned)((n + 255) / 256)), 256, 0, stream>>>(p, v, (int)n);
  };
  auto mv = [&](const float* A, const float* B, const float* bias, float* C,
                int n, int k, int f, int act) {
    size_t tot = (size_t)n * f;
    matvec_act<<<dim3((unsigned)((tot + 127) / 128)), 128, 0, stream>>>(A, B, bias, C, n, k, f, act);
  };
  auto pack = [&](const float* B, bf16* P, int K, int Hc, int Hp, int Kc) {
    long tot = (long)Kc * Hp * 32;
    packB<<<dim3((unsigned)((tot + 255) / 256)), 256, 0, stream>>>(B, P, K, Hc, Hp, Kc);
  };
  auto gat = [&](const float* z, int F, const float* as, const float* ad,
                 float* outp, int act) {
    gat_coef<<<dim3((N + 63) / 64), 64, 0, stream>>>(z, as, ad, sb, db, N, F);
    fill(mb, -3.0e38f, N);
    fill(denb, 0.f, N);
    fill(numb, 0.f, (size_t)N * F);
    gat_edge_max<<<dim3((Etot + 127) / 128), 128, 0, stream>>>(ei, E, Etot, sb, db, eb, mb);
    gat_edge_w<<<dim3((Etot + 127) / 128), 128, 0, stream>>>(ei, E, Etot, eb, mb, wb, denb);
    gat_edge_acc<<<dim3(Etot), 64, 0, stream>>>(ei, E, wb, z, numb, F);
    gat_final<<<dim3((unsigned)(((size_t)N * F + 127) / 128)), 128, 0, stream>>>(numb, denb, outp, N, F, act);
  };

  fill(scal, 0.f, 16);

  // Pack weights to bf16 WMMA-fragment order (tiny vs the 49MB y stream)
  pack(gat_W1, pB1, Dd, H, H, Kc1);
  pack(enc_W1, pB2, Dd, H, H, Kc1);
  pack(dec_W2, pBd, DH, Dd, Dpad, KcD);

  // Fused big GEMMs: y streamed once -> z1 = y@gat_W1, hh = elu(y@enc_W1+b1)
  gemm_dualB_wmma<<<dim3(N / 16, H / 32), 32, 0, stream>>>(
      y, pB1, pB2, enc_b1, z1, hh, N, Dd, H, Kc1);

  // GAT layer 1 (F=H, elu) -> hbuf
  gat(z1, H, gat_a1s, gat_a1d, hbuf, 1);

  // Small matvecs
  mv(hbuf, gat_Wmu,  nullptr, zmu,  N, H, GZ, 0);
  mv(hbuf, gat_Wvar, nullptr, zvar, N, H, GZ, 0);
  mv(hh,   enc_Wmu,  enc_bmu,  qmu,  N, H, LZ, 0);
  mv(hh,   enc_Wvar, enc_bvar, qvar, N, H, LZ, 2);

  // GAT mu / var layers (F=GZ)
  gat(zmu,  GZ, gat_ams, gat_amd, gmu,  0);
  gat(zvar, GZ, gat_avs, gat_avd, gvar, 2);

  gnn_sample<<<dim3((N * GZ + 127) / 128), 128, 0, stream>>>(
      gmu, gvar, latent, gnn_out, &scal[14], N, LZ, GZ);

  // SVGP kernel matrices
  kmm_build<<<dim3((64 * 64 + 127) / 128), 128, 0, stream>>>(Zp, Kmm);
  invert64_logdet<<<1, 128, 0, stream>>>(Kmm, Kinv, &scal[0]);
  knm_build<<<dim3((N * 64 + 127) / 128), 128, 0, stream>>>(x, Zp, Knm, N);

  for (int l = 0; l < LZ; l++) {
    rhs_build<<<1, 64, 0, stream>>>(Knm, qmu, qvar, rhs, N, l, LZ);
    sigma_build<<<dim3(64), 64, 0, stream>>>(Kmm, Knm, qvar, sig, N, l, LZ);
    invert64_logdet<<<1, 128, 0, stream>>>(sig, sigi, &scal[1]);
    matvec64<<<1, 64, 0, stream>>>(sigi, rhs, tmp1);
    matvec64<<<1, 64, 0, stream>>>(Kmm, tmp1, muh);
    kl_kernel<<<1, 64, 0, stream>>>(sigi, Kmm, muh, tmp1, &scal[1], &scal[0], &scal[2 + l]);
    svgp_pern<<<dim3(N), 64, 0, stream>>>(Knm, sigi, Kinv, tmp1, qmu, qvar, l, LZ,
                                          pm, pv, &scal[7 + l]);
  }

  ce_sample<<<dim3((N * LZ + 127) / 128), 128, 0, stream>>>(
      pm, pv, qmu, qvar, latent, svgp_out, &scal[12], N, LZ, GZ);
  align_kernel<<<dim3((N + 127) / 128), 128, 0, stream>>>(
      svgp_out, gnn_out, &scal[15], N, LZ, GZ);

  // Decoder: hd = elu(latent@dec_W1 + b1), then fused WMMA GEMM + loss
  mv(latent, dec_W1, dec_b1, hd, N, LZ + GZ, DH, 1);
  gemm_dec_wmma<<<dim3(N / 16, (Dd + 15) / 16), 32, 0, stream>>>(
      hd, pBd, dec_b2, y, &scal[13], N, DH, Dd, Dpad);

  finalize<<<1, 1, 0, stream>>>(scal, bptr, out, Dd, LZ, GZ);
}
